// CosineSimilarityCodebook_29463475651258
// MI455X (gfx1250) — matrix operand order repaired and checked by reading, not compiled, not run on hardware
//
#include <hip/hip_runtime.h>
#include <hip/hip_bf16.h>
#include <math.h>

typedef __attribute__((ext_vector_type(16))) _Float16 v16h;
typedef __attribute__((ext_vector_type(8)))  _Float16 v8h;
typedef __attribute__((ext_vector_type(8)))  float    v8f;

#define DIM 256
#define LDS_STRIDE 264   // 256 halfs + 8-half pad (16B): 528B row stride, bank-friendly
#define KSPLITS 4        // codebook split factor (parallelism in K dimension)

__device__ inline float wave_reduce_sum_f32(float v) {
    #pragma unroll
    for (int off = 16; off > 0; off >>= 1)
        v += __shfl_xor(v, off, 32);
    return v;
}

// One wave (32 lanes) l2-normalizes one 256-float row and writes it as f16.
__global__ __launch_bounds__(256) void l2norm_to_f16_kernel(
    const float* __restrict__ src, _Float16* __restrict__ dst, int nrows) {
    const int wave = (int)((blockIdx.x * blockDim.x + threadIdx.x) >> 5);
    const int lane = (int)(threadIdx.x & 31);
    if (wave >= nrows) return;

    const float4* s = (const float4*)(src + (size_t)wave * DIM);
    float4 a = s[lane * 2 + 0];
    float4 b = s[lane * 2 + 1];
    float ss = a.x*a.x + a.y*a.y + a.z*a.z + a.w*a.w
             + b.x*b.x + b.y*b.y + b.z*b.z + b.w*b.w;
    ss = wave_reduce_sum_f32(ss);
    const float scale = 1.0f / fmaxf(sqrtf(ss), 1e-12f);

    v8h o;
    o[0] = (_Float16)(a.x * scale); o[1] = (_Float16)(a.y * scale);
    o[2] = (_Float16)(a.z * scale); o[3] = (_Float16)(a.w * scale);
    o[4] = (_Float16)(b.x * scale); o[5] = (_Float16)(b.y * scale);
    o[6] = (_Float16)(b.z * scale); o[7] = (_Float16)(b.w * scale);
    *(v8h*)(dst + (size_t)wave * DIM + lane * 8) = o;
}

// Fused cosine-sim GEMM (f16 WMMA, f32 accum) + partial argmax over a K-split.
// 8 waves / block, each owns a 16-row M-tile with A resident in registers.
// B (codebook) chunks of 32 columns are staged in LDS, double-buffered,
// shared by all 8 waves. blockIdx.y selects the K-split.
__global__ __launch_bounds__(256) void cosine_argmax_partial_kernel(
    const _Float16* __restrict__ xn,     // [N, 256] normalized f16
    const _Float16* __restrict__ embn,   // [K, 256] normalized f16
    float* __restrict__ pval,            // [KSPLITS, N] partial best value
    int*   __restrict__ pidx,            // [KSPLITS, N] partial best index
    int nrows, int chunks_per_split) {
    __shared__ __align__(16) _Float16 smem[2 * 32 * LDS_STRIDE];

    const int lane = (int)(threadIdx.x & 31);
    const int wid  = (int)(threadIdx.x >> 5);
    const int row_base = ((int)blockIdx.x * 8 + wid) * 16;
    const int split = (int)blockIdx.y;
    const int hl = lane >> 4;    // lane half (0: lanes 0-15, 1: lanes 16-31)
    const int l  = lane & 15;

    // Staging role: 256 threads move 32 rows x 512B; 8 threads per row, 64B each.
    const int srow  = (int)(threadIdx.x >> 3);   // 0..31 (column within chunk)
    const int spart = (int)(threadIdx.x & 7);    // 0..7  (32-half slice of row)

    const int cb0 = split * chunks_per_split;
    const int cb_end = cb0 + chunks_per_split;

    // ---- Load A tile (16 rows x 256 K) into 8 x v16h (64 VGPRs). ----
    // A k order used here (contiguous; B loads match): half 0 holds K {k0..k0+15} of row l,
    //                        half 1 holds K {k0+16..k0+31}.
    v16h A[8];
    {
        const _Float16* arow = xn + (size_t)(row_base + l) * DIM;
        #pragma unroll
        for (int kk = 0; kk < 8; ++kk) {
            const int k0 = kk * 32;
            v8h lo = *(const v8h*)(arow + k0 + hl * 16);
            v8h hi = *(const v8h*)(arow + k0 + hl * 16 + 8);
            A[kk] = __builtin_shufflevector(lo, hi,
                        0,1,2,3,4,5,6,7,8,9,10,11,12,13,14,15);
        }
    }

    float best[8];
    int   bidx[8];
    #pragma unroll
    for (int s = 0; s < 8; ++s) { best[s] = -INFINITY; bidx[s] = 0; }

    // ---- Prologue: stage first chunk of this split into LDS buffer 0. ----
    {
        const uint4* g = (const uint4*)
            (embn + (size_t)(cb0 * 32 + srow) * DIM + spart * 32);
        uint4 st0 = g[0], st1 = g[1], st2 = g[2], st3 = g[3];
        uint4* d = (uint4*)(smem + srow * LDS_STRIDE + spart * 32);
        d[0] = st0; d[1] = st1; d[2] = st2; d[3] = st3;
    }
    __syncthreads();

    for (int cb = cb0; cb < cb_end; ++cb) {
        // (a) Issue next chunk's global loads early; they complete under WMMA.
        uint4 st0, st1, st2, st3;
        const bool have_next = (cb + 1) < cb_end;
        if (have_next) {
            const uint4* g = (const uint4*)
                (embn + (size_t)((cb + 1) * 32 + srow) * DIM + spart * 32);
            st0 = g[0]; st1 = g[1]; st2 = g[2]; st3 = g[3];
            if (cb + 2 < cb_end)
                __builtin_prefetch(
                    embn + (size_t)((cb + 2) * 32 + srow) * DIM + spart * 32, 0, 0);
        }

        // (b) 16-WMMA chain off the current LDS buffer, software-pipelined B:
        //     loads for step kk+1 are in flight across the WMMAs of step kk.
        const _Float16* bb = smem + ((cb - cb0) & 1) * (32 * LDS_STRIDE);
        const _Float16* b0p = bb + l * LDS_STRIDE + hl * 16;          // cols cb*32+l
        const _Float16* b1p = bb + (l + 16) * LDS_STRIDE + hl * 16;   // cols cb*32+16+l

        v8f c0 = {};
        v8f c1 = {};
        v16h b0c, b1c, b0n, b1n;
        {
            v8h lo0 = *(const v8h*)(b0p);
            v8h hi0 = *(const v8h*)(b0p + 8);
            v8h lo1 = *(const v8h*)(b1p);
            v8h hi1 = *(const v8h*)(b1p + 8);
            b0c = __builtin_shufflevector(lo0, hi0,
                      0,1,2,3,4,5,6,7,8,9,10,11,12,13,14,15);
            b1c = __builtin_shufflevector(lo1, hi1,
                      0,1,2,3,4,5,6,7,8,9,10,11,12,13,14,15);
        }
        #pragma unroll
        for (int kk = 0; kk < 8; ++kk) {
            if (kk < 7) {
                v8h lo0 = *(const v8h*)(b0p + (kk + 1) * 32);
                v8h hi0 = *(const v8h*)(b0p + (kk + 1) * 32 + 8);
                v8h lo1 = *(const v8h*)(b1p + (kk + 1) * 32);
                v8h hi1 = *(const v8h*)(b1p + (kk + 1) * 32 + 8);
                b0n = __builtin_shufflevector(lo0, hi0,
                          0,1,2,3,4,5,6,7,8,9,10,11,12,13,14,15);
                b1n = __builtin_shufflevector(lo1, hi1,
                          0,1,2,3,4,5,6,7,8,9,10,11,12,13,14,15);
            }
            c0 = __builtin_amdgcn_wmma_f32_16x16x32_f16(
                     false, A[kk], false, b0c, (short)0, c0, false, false);
            c1 = __builtin_amdgcn_wmma_f32_16x16x32_f16(
                     false, A[kk], false, b1c, (short)0, c1, false, false);
            b0c = b0n; b1c = b1n;
        }

        // c[s] = score[row_base + s + hl*8][col]; fold into running argmax.
        const int col0 = cb * 32 + l;
        #pragma unroll
        for (int s = 0; s < 8; ++s) {
            if (c0[s] > best[s]) { best[s] = c0[s]; bidx[s] = col0; }
            if (c1[s] > best[s]) { best[s] = c1[s]; bidx[s] = col0 + 16; }
        }

        // (c) Commit staged registers into the other LDS buffer.
        if (have_next) {
            uint4* d = (uint4*)(smem + ((cb - cb0 + 1) & 1) * (32 * LDS_STRIDE)
                                + srow * LDS_STRIDE + spart * 32);
            d[0] = st0; d[1] = st1; d[2] = st2; d[3] = st3;
        }
        // (d) One barrier per iteration covers both read-done and write-done.
        __syncthreads();
    }

    // ---- Cross-lane argmax within each 16-lane half (xor butterfly). ----
    #pragma unroll
    for (int s = 0; s < 8; ++s) {
        #pragma unroll
        for (int off = 8; off >= 1; off >>= 1) {
            float ov = __shfl_xor(best[s], off, 32);
            int   oi = __shfl_xor(bidx[s], off, 32);
            if (ov > best[s] || (ov == best[s] && oi < bidx[s])) {
                best[s] = ov; bidx[s] = oi;
            }
        }
    }
    // Lanes with l==0 (lanes 0 and 16) hold rows row_base+hl*8 .. +7.

    if (l == 0) {
        const size_t base = (size_t)split * nrows + row_base + hl * 8;
        #pragma unroll
        for (int s = 0; s < 8; ++s) {
            pval[base + s] = best[s];
            pidx[base + s] = bidx[s];
        }
    }
}

// One wave per row: combine K-split partials (ascending split order keeps
// first-max semantics), gather the UN-normalized codebook row, write index.
__global__ __launch_bounds__(256) void combine_gather_kernel(
    const float* __restrict__ pval, const int* __restrict__ pidx,
    const float* __restrict__ emb,
    float* __restrict__ out_q, float* __restrict__ out_i, int nrows) {
    const int row  = (int)((blockIdx.x * blockDim.x + threadIdx.x) >> 5);
    const int lane = (int)(threadIdx.x & 31);
    if (row >= nrows) return;

    // All lanes compute the winner redundantly (KSPLITS reads, cheap).
    float bv = -INFINITY;
    int   bi = 0;
    #pragma unroll
    for (int s = 0; s < KSPLITS; ++s) {
        const float v = pval[(size_t)s * nrows + row];
        const int   i = pidx[(size_t)s * nrows + row];
        if (v > bv) { bv = v; bi = i; }
    }

    const float4* srcv = (const float4*)(emb + (size_t)bi * DIM);
    float4* dstv = (float4*)(out_q + (size_t)row * DIM);
    dstv[lane]      = srcv[lane];
    dstv[lane + 32] = srcv[lane + 32];
    if (lane == 0) out_i[row] = (float)bi;
}

extern "C" void kernel_launch(void* const* d_in, const int* in_sizes, int n_in,
                              void* d_out, int out_size, void* d_ws, size_t ws_size,
                              hipStream_t stream) {
    const float* x   = (const float*)d_in[0];   // [4,4096,256] f32
    const float* emb = (const float*)d_in[1];   // [8192,256]  f32
    const int N = in_sizes[0] / DIM;            // 16384
    const int K = in_sizes[1] / DIM;            // 8192

    _Float16* xn   = (_Float16*)d_ws;                            // 8 MB
    _Float16* embn = xn + (size_t)N * DIM;                       // 4 MB
    float*    pval = (float*)(embn + (size_t)K * DIM);           // 256 KB
    int*      pidx = (int*)(pval + (size_t)KSPLITS * N);         // 256 KB
    float* out_q = (float*)d_out;
    float* out_i = out_q + (size_t)N * DIM;

    // 8 rows (waves) per 256-thread block.
    l2norm_to_f16_kernel<<<(N + 7) / 8, 256, 0, stream>>>(x, xn, N);
    l2norm_to_f16_kernel<<<(K + 7) / 8, 256, 0, stream>>>(emb, embn, K);

    // 8 M-tiles per block, K split 4 ways: 512 blocks / 4096 matrix waves.
    const int chunks_per_split = (K / 32) / KSPLITS;   // 64
    dim3 grid(N / 128, KSPLITS);
    cosine_argmax_partial_kernel<<<grid, 256, 0, stream>>>(
        xn, embn, pval, pidx, N, chunks_per_split);

    combine_gather_kernel<<<(N + 7) / 8, 256, 0, stream>>>(
        pval, pidx, emb, out_q, out_i, N);
}
